// ContrastiveLossWithFreeEnergy_32212254720563
// MI455X (gfx1250) — matrix operand order repaired and checked by reading, not compiled, not run on hardware
//
#include <hip/hip_runtime.h>
#include <hip/hip_bf16.h>
#include <math.h>

typedef __bf16 bf16;
typedef bf16  v16bf __attribute__((ext_vector_type(16)));
typedef float v8f   __attribute__((ext_vector_type(8)));

#define WMMA_BF16(a,b,c) __builtin_amdgcn_wmma_f32_16x16x32_bf16(false,(a),false,(b),(short)0,(c),false,false)

// ---------------- reduction helpers ----------------
__device__ __forceinline__ float block_sum(float v, float* buf) {
    int t = threadIdx.x;
    buf[t] = v; __syncthreads();
    for (int off = blockDim.x >> 1; off > 0; off >>= 1) {
        if (t < off) buf[t] += buf[t + off];
        __syncthreads();
    }
    float r = buf[0]; __syncthreads();
    return r;
}
__device__ __forceinline__ float block_min(float v, float* buf) {
    int t = threadIdx.x;
    buf[t] = v; __syncthreads();
    for (int off = blockDim.x >> 1; off > 0; off >>= 1) {
        if (t < off) buf[t] = fminf(buf[t], buf[t + off]);
        __syncthreads();
    }
    float r = buf[0]; __syncthreads();
    return r;
}
__device__ __forceinline__ float block_max(float v, float* buf) {
    int t = threadIdx.x;
    buf[t] = v; __syncthreads();
    for (int off = blockDim.x >> 1; off > 0; off >>= 1) {
        if (t < off) buf[t] = fmaxf(buf[t], buf[t + off]);
        __syncthreads();
    }
    float r = buf[0]; __syncthreads();
    return r;
}

// ---------------- init ----------------
__global__ void zero_hist_kernel(unsigned* hist) {
    hist[threadIdx.x] = 0u;   // <<<1,512>>>
}

// ---------------- weight conversion: W3c^T and W4^T to bf16 [N][K] ----------------
__global__ void conv_weights_kernel(const float* __restrict__ W3,
                                    const float* __restrict__ W4,
                                    bf16* __restrict__ W3ct,
                                    bf16* __restrict__ W4t) {
    int idx = blockIdx.x * blockDim.x + threadIdx.x;   // 98304 threads
    if (idx < 65536) {
        int n = idx >> 8, k = idx & 255;
        W3ct[n * 256 + k] = (bf16)W3[(512 + k) * 256 + n];
    } else {
        int j = idx - 65536;                            // 32768
        int n = j >> 8, k = j & 255;
        W4t[n * 256 + k] = (bf16)W4[k * 128 + n];
    }
}

// ---------------- encoder: e = relu(LN(x@W1+b1)*g1+be1)@W2+b2 ----------------
__global__ __launch_bounds__(256)
void encoder_kernel(const float* __restrict__ x,   // [256,512]
                    const float* __restrict__ W1, const float* __restrict__ b1,
                    const float* __restrict__ g1, const float* __restrict__ be1,
                    const float* __restrict__ W2, const float* __restrict__ b2,
                    float* __restrict__ ef, bf16* __restrict__ eb) {
    __shared__ float sx[512];
    __shared__ float sr[256];
    __shared__ float buf[256];
    int row = blockIdx.x, t = threadIdx.x;
    sx[t] = x[row * 512 + t];
    sx[t + 256] = x[row * 512 + 256 + t];
    __syncthreads();
    float acc = b1[t];
    for (int k = 0; k < 512; ++k) acc += sx[k] * W1[k * 256 + t];
    float s  = block_sum(acc, buf);
    float sq = block_sum(acc * acc, buf);
    float mu = s / 256.0f;
    float var = sq / 256.0f - mu * mu;
    float h = (acc - mu) * rsqrtf(var + 1e-5f) * g1[t] + be1[t];
    sr[t] = fmaxf(h, 0.0f);
    __syncthreads();
    float o = b2[t];
    for (int k = 0; k < 256; ++k) o += sr[k] * W2[k * 256 + t];
    ef[row * 256 + t] = o;
    eb[row * 256 + t] = (bf16)o;
}

// ---------------- P1 = e1@W3a + b3 ; P2 = e2@W3b ----------------
__global__ __launch_bounds__(256)
void pmat_kernel(const float* __restrict__ e1, const float* __restrict__ e2,
                 const float* __restrict__ W3, const float* __restrict__ b3,
                 float* __restrict__ P1, float* __restrict__ P2) {
    __shared__ float sx[256];
    int which = blockIdx.x >> 8, row = blockIdx.x & 255, t = threadIdx.x;
    const float* e = which ? e2 : e1;
    const float* W = W3 + which * 65536;    // rows 0..255 or 256..511
    float* out = which ? P2 : P1;
    sx[t] = e[row * 256 + t];
    __syncthreads();
    float acc = which ? 0.0f : b3[t];
    for (int k = 0; k < 256; ++k) acc += sx[k] * W[k * 256 + t];
    out[row * 256 + t] = acc;
}

// ---------------- inverse norms for cosine sim ----------------
__global__ __launch_bounds__(256)
void invnorm_kernel(const float* __restrict__ fo, const float* __restrict__ fr,
                    float* __restrict__ invn) {
    __shared__ float buf[256];
    int which = blockIdx.x >> 8, row = blockIdx.x & 255, t = threadIdx.x;
    const float* f = which ? fr : fo;
    float a = f[row * 512 + t], b = f[row * 512 + 256 + t];
    float s = block_sum(a * a + b * b, buf);
    if (t == 0) invn[blockIdx.x] = rsqrtf(s);
}

// ---------------- top-5 neighbor jaccard ----------------
__global__ __launch_bounds__(256)
void topk_kernel(const float* __restrict__ fo, const float* __restrict__ fr,
                 const float* __restrict__ invn, float* __restrict__ jacc) {
    __shared__ float srow[512];
    __shared__ float sim[256];
    __shared__ float rv[256];
    __shared__ int   ri[256];
    __shared__ int   topO[5], topR[5];
    int i = blockIdx.x, t = threadIdx.x;
    for (int pass = 0; pass < 2; ++pass) {
        const float* f  = pass ? fr : fo;
        const float* in = invn + pass * 256;
        srow[t] = f[i * 512 + t];
        srow[t + 256] = f[i * 512 + 256 + t];
        __syncthreads();
        float d = 0.0f;
        for (int k = 0; k < 512; ++k) d += srow[k] * f[t * 512 + k];
        float v = d * in[i] * in[t];
        sim[t] = (t == i) ? -1e30f : v;
        __syncthreads();
        for (int s = 0; s < 5; ++s) {
            rv[t] = sim[t]; ri[t] = t;
            __syncthreads();
            for (int off = 128; off > 0; off >>= 1) {
                if (t < off) {
                    float vo = rv[t + off]; int io = ri[t + off];
                    if (vo > rv[t] || (vo == rv[t] && io < ri[t])) { rv[t] = vo; ri[t] = io; }
                }
                __syncthreads();
            }
            if (t == 0) {
                int w = ri[0];
                if (pass == 0) topO[s] = w; else topR[s] = w;
                sim[w] = -1e30f;
            }
            __syncthreads();
        }
        __syncthreads();
    }
    if (t == 0) {
        int inter = 0;
        for (int a = 0; a < 5; ++a)
            for (int b = 0; b < 5; ++b)
                if (topO[a] == topR[b]) inter++;
        float uni = 10.0f - (float)inter;
        jacc[i] = (float)inter / (uni + 1e-8f);
    }
}

__global__ __launch_bounds__(256)
void jacc_final_kernel(const float* __restrict__ jacc, float* __restrict__ scal) {
    __shared__ float buf[256];
    float s = block_sum(jacc[threadIdx.x], buf);
    if (threadIdx.x == 0) scal[10] = s / 256.0f;   // mean jaccard
}

// ---------------- reconstruction MSE ----------------
__global__ __launch_bounds__(256)
void recon_part_kernel(const float* __restrict__ o, const float* __restrict__ im,
                       int n, float* __restrict__ part) {
    __shared__ float buf[256];
    float s = 0.0f;
    for (int idx = blockIdx.x * blockDim.x + threadIdx.x; idx < n;
         idx += blockDim.x * gridDim.x) {
        float d = o[idx] - im[idx];
        s += d * d;
    }
    s = block_sum(s, buf);
    if (threadIdx.x == 0) part[blockIdx.x] = s;
}

__global__ __launch_bounds__(512)
void recon_final_kernel(const float* __restrict__ part, float* __restrict__ scal) {
    __shared__ float buf[512];
    float s = block_sum(part[threadIdx.x], buf);
    if (threadIdx.x == 0) scal[0] = s / 3145728.0f;
}

// ---------------- cf stats: sum|x|, min, max ----------------
__global__ __launch_bounds__(256)
void stats_part_kernel(const float* __restrict__ x, int n,
                       float* __restrict__ pa, float* __restrict__ pmn,
                       float* __restrict__ pmx) {
    __shared__ float buf[256];
    float sa = 0.0f, mn = 3.4e38f, mx = -3.4e38f;
    for (int idx = blockIdx.x * blockDim.x + threadIdx.x; idx < n;
         idx += blockDim.x * gridDim.x) {
        float v = x[idx];
        sa += fabsf(v);
        mn = fminf(mn, v);
        mx = fmaxf(mx, v);
    }
    sa = block_sum(sa, buf);
    mn = block_min(mn, buf);
    mx = block_max(mx, buf);
    if (threadIdx.x == 0) { pa[blockIdx.x] = sa; pmn[blockIdx.x] = mn; pmx[blockIdx.x] = mx; }
}

__global__ __launch_bounds__(512)
void stats_final_kernel(const float* __restrict__ pa, const float* __restrict__ pmn,
                        const float* __restrict__ pmx, float* __restrict__ scal,
                        int sa_slot, int mn_slot, int mx_slot) {
    __shared__ float buf[512];
    float sa = block_sum(pa[threadIdx.x], buf);
    float mn = block_min(pmn[threadIdx.x], buf);
    float mx = block_max(pmx[threadIdx.x], buf);
    if (threadIdx.x == 0) { scal[sa_slot] = sa; scal[mn_slot] = mn; scal[mx_slot] = mx; }
}

// ---------------- histogram + entropy ----------------
__global__ __launch_bounds__(256)
void hist_kernel(const float* __restrict__ x, int n, const float* __restrict__ scal,
                 int mn_slot, int mx_slot, unsigned* __restrict__ hist) {
    __shared__ unsigned hl[256];
    hl[threadIdx.x] = 0u;
    __syncthreads();
    float mn = scal[mn_slot], mx = scal[mx_slot];
    float inv = 1.0f / (mx - mn + 1e-8f);
    for (int idx = blockIdx.x * blockDim.x + threadIdx.x; idx < n;
         idx += blockDim.x * gridDim.x) {
        float xn = (x[idx] - mn) * inv;
        int b = (int)floorf(xn * 256.0f);
        b = min(max(b, 0), 255);
        atomicAdd(&hl[b], 1u);
    }
    __syncthreads();
    atomicAdd(&hist[threadIdx.x], hl[threadIdx.x]);
}

__global__ __launch_bounds__(256)
void entropy_kernel(const unsigned* __restrict__ hist, float total,
                    float* __restrict__ scal, int slot) {
    __shared__ float buf[256];
    float p = (float)hist[threadIdx.x] / total;
    float e = -p * log2f(p + 1e-8f);
    float s = block_sum(e, buf);
    if (threadIdx.x == 0) scal[slot] = s;
}

// ---------------- energy matrix: WMMA bf16 MLP over all (i,j) pairs ----------------
// grid 256 = (bi,bj) 16x16 tiles of E; block = 128 threads = 4 waves.
// Each wave handles i-rows bi*16+{wave,wave+4,wave+8,wave+12}, 16 j's each.
// N-tiles processed in PAIRS with two independent accumulators to break the
// WMMA->WMMA accumulator RAW chain (each A-fragment used twice back-to-back).
__global__ __launch_bounds__(128)
void energy_kernel(const bf16* __restrict__ e1b, const bf16* __restrict__ e2b,
                   const float* __restrict__ P1, const float* __restrict__ P2,
                   const bf16* __restrict__ W3ct, const bf16* __restrict__ W4t,
                   const float* __restrict__ g3, const float* __restrict__ be3,
                   const float* __restrict__ b4, const float* __restrict__ W5,
                   const float* __restrict__ b5, float* __restrict__ E) {
    __shared__ bf16  sE2[16][256];        // 8 KB : e2 rows of this j-tile
    __shared__ float sP2[16][256];        // 16 KB: P2 rows of this j-tile
    __shared__ bf16  sPre[4][16][256];    // 32 KB: per-wave pre-LN (biased) rows
    __shared__ float sMu[4][16], sRs[4][16];
    __shared__ float sg3[256], sbe3[256];
    __shared__ float sb4[128], sw5[128];

    const int bi = blockIdx.x >> 4, bj = blockIdx.x & 15;
    const int tid = threadIdx.x;
    const int wave = tid >> 5, lane = tid & 31;
    const int l = lane & 15, h = lane >> 4;

    for (int idx = tid; idx < 16 * 256; idx += 128) {
        int r = idx >> 8, c = idx & 255;
        sE2[r][c] = e2b[(bj * 16 + r) * 256 + c];
        sP2[r][c] = P2[(bj * 16 + r) * 256 + c];
    }
    for (int idx = tid; idx < 256; idx += 128) { sg3[idx] = g3[idx]; sbe3[idx] = be3[idx]; }
    if (tid < 128) { sb4[tid] = b4[tid]; sw5[tid] = W5[tid]; }
    __syncthreads();

    const float b5v = b5[0];

    for (int mt = wave; mt < 16; mt += 4) {
        const int i = bi * 16 + mt;
        const bf16*  e1row = e1b + i * 256;   // wave-uniform
        const float* p1row = P1  + i * 256;   // wave-uniform

        // ---- build all 8 K-tile A-fragments of (e1_i * e2_j), 16-bit A 16x32 layout
        v16bf A3[8];
        #pragma unroll
        for (int kt = 0; kt < 8; ++kt) {
            const int kb = kt * 32;
            #pragma unroll
            for (int e = 0; e < 16; ++e) {
                int k = kb + h * 16 + e;
                A3[kt][e] = (bf16)((float)e1row[k] * (float)sE2[l][k]);
            }
        }

        // ---- layer 3 GEMM, two N-tiles at a time; park pre-LN rows in LDS ----
        float sum[8], sqs[8];
        #pragma unroll
        for (int v = 0; v < 8; ++v) { sum[v] = 0.0f; sqs[v] = 0.0f; }

        for (int nt = 0; nt < 16; nt += 2) {
            v8f acc0, acc1;
            #pragma unroll
            for (int v = 0; v < 8; ++v) { acc0[v] = 0.0f; acc1[v] = 0.0f; }
            #pragma unroll
            for (int kt = 0; kt < 8; ++kt) {
                const bf16* bp0 = W3ct + ((nt + 0) * 16 + l) * 256 + kt * 32 + h * 16;
                const bf16* bp1 = W3ct + ((nt + 1) * 16 + l) * 256 + kt * 32 + h * 16;
                v16bf b0 = *(const v16bf*)bp0;
                v16bf b1 = *(const v16bf*)bp1;
                acc0 = WMMA_BF16(A3[kt], b0, acc0);
                acc1 = WMMA_BF16(A3[kt], b1, acc1);
            }
            const int n0 = 16 * (nt + 0) + l;
            const int n1 = 16 * (nt + 1) + l;
            float p1v0 = p1row[n0];
            float p1v1 = p1row[n1];
            #pragma unroll
            for (int v = 0; v < 8; ++v) {
                float x0 = acc0[v] + p1v0 + sP2[v + 8 * h][n0];
                float x1 = acc1[v] + p1v1 + sP2[v + 8 * h][n1];
                sum[v] += x0 + x1;
                sqs[v] += x0 * x0 + x1 * x1;
                sPre[wave][v + 8 * h][n0] = (bf16)x0;
                sPre[wave][v + 8 * h][n1] = (bf16)x1;
            }
        }

        // ---- rowwise LayerNorm stats (rows live in 16-lane halves) ----
        #pragma unroll
        for (int v = 0; v < 8; ++v) {
            #pragma unroll
            for (int m = 1; m < 16; m <<= 1) {
                sum[v] += __shfl_xor(sum[v], m, 16);
                sqs[v] += __shfl_xor(sqs[v], m, 16);
            }
        }
        if (l == 0) {
            #pragma unroll
            for (int v = 0; v < 8; ++v) {
                float mu = sum[v] / 256.0f;
                float var = sqs[v] / 256.0f - mu * mu;
                sMu[wave][v + 8 * h] = mu;
                sRs[wave][v + 8 * h] = rsqrtf(var + 1e-5f);
            }
        }

        // ---- layer 4 A-fragments: LN + relu applied inline from sPre ----
        const float muL = sMu[wave][l];
        const float rsL = sRs[wave][l];
        v16bf A4[8];
        #pragma unroll
        for (int kt = 0; kt < 8; ++kt) {
            const int kb = kt * 32;
            #pragma unroll
            for (int e = 0; e < 16; ++e) {
                int k = kb + h * 16 + e;
                float x = (float)sPre[wave][l][k];
                x = (x - muL) * rsL * sg3[k] + sbe3[k];
                A4[kt][e] = (bf16)fmaxf(x, 0.0f);
            }
        }

        // ---- layer 4 GEMM (paired N-tiles) + layer 5 fused ----
        float p[8];
        #pragma unroll
        for (int v = 0; v < 8; ++v) p[v] = 0.0f;
        for (int nt = 0; nt < 8; nt += 2) {
            v8f acc0, acc1;
            #pragma unroll
            for (int v = 0; v < 8; ++v) { acc0[v] = 0.0f; acc1[v] = 0.0f; }
            #pragma unroll
            for (int kt = 0; kt < 8; ++kt) {
                const bf16* bp0 = W4t + ((nt + 0) * 16 + l) * 256 + kt * 32 + h * 16;
                const bf16* bp1 = W4t + ((nt + 1) * 16 + l) * 256 + kt * 32 + h * 16;
                v16bf b0 = *(const v16bf*)bp0;
                v16bf b1 = *(const v16bf*)bp1;
                acc0 = WMMA_BF16(A4[kt], b0, acc0);
                acc1 = WMMA_BF16(A4[kt], b1, acc1);
            }
            const int n0 = 16 * (nt + 0) + l;
            const int n1 = 16 * (nt + 1) + l;
            float b40 = sb4[n0], w50 = sw5[n0];
            float b41 = sb4[n1], w51 = sw5[n1];
            #pragma unroll
            for (int v = 0; v < 8; ++v) {
                p[v] += fmaxf(acc0[v] + b40, 0.0f) * w50;
                p[v] += fmaxf(acc1[v] + b41, 0.0f) * w51;
            }
        }
        #pragma unroll
        for (int v = 0; v < 8; ++v) {
            #pragma unroll
            for (int m = 1; m < 16; m <<= 1) p[v] += __shfl_xor(p[v], m, 16);
        }
        if (l == 0) {
            #pragma unroll
            for (int v = 0; v < 8; ++v) {
                int jj = v + 8 * h;
                E[i * 256 + bj * 16 + jj] = p[v] + b5v;
            }
        }
    }
}

// ---------------- margin loss ----------------
__global__ __launch_bounds__(256)
void margin_part_kernel(const float* __restrict__ E, float* __restrict__ part) {
    __shared__ float buf[256];
    int i = blockIdx.x, j = threadIdx.x;
    float pos = E[i * 257];
    float v = (i == j) ? 0.0f : fmaxf(0.0f, 1.0f + E[i * 256 + j] - pos);
    float s = block_sum(v, buf);
    if (j == 0) part[i] = s;
}

__global__ __launch_bounds__(256)
void margin_final_kernel(const float* __restrict__ part, float* __restrict__ scal) {
    __shared__ float buf[256];
    float s = block_sum(part[threadIdx.x], buf);
    if (threadIdx.x == 0) scal[9] = s / (256.0f * 255.0f);
}

// ---------------- final combine ----------------
__global__ void combine_kernel(const float* __restrict__ scal, float* __restrict__ out) {
    float recon = scal[0];
    float sp = 0.5f * (scal[1] / 4194304.0f + scal[2] / 2097152.0f);
    float en = 0.5f * (scal[7] + scal[8]);
    float compress = sp + 0.1f * en;                        // BETA
    float energy = scal[9] + 0.3f * (1.0f - scal[10]);
    float total = recon + 0.01f * compress + 0.1f * energy; // LAM_C, LAM_E
    out[0] = total; out[1] = recon; out[2] = energy; out[3] = compress;
}

// ---------------- launch ----------------
extern "C" void kernel_launch(void* const* d_in, const int* in_sizes, int n_in,
                              void* d_out, int out_size, void* d_ws, size_t ws_size,
                              hipStream_t stream) {
    const float* outputs = (const float*)d_in[0];
    const float* images  = (const float*)d_in[1];
    const float* f_orig  = (const float*)d_in[2];
    const float* f_recon = (const float*)d_in[3];
    const float* cf1     = (const float*)d_in[4];
    const float* cf2     = (const float*)d_in[5];
    const float* W1  = (const float*)d_in[6];
    const float* b1  = (const float*)d_in[7];
    const float* g1  = (const float*)d_in[8];
    const float* be1 = (const float*)d_in[9];
    const float* W2  = (const float*)d_in[10];
    const float* b2  = (const float*)d_in[11];
    const float* W3  = (const float*)d_in[12];
    const float* b3  = (const float*)d_in[13];
    const float* g3  = (const float*)d_in[14];
    const float* be3 = (const float*)d_in[15];
    const float* W4  = (const float*)d_in[16];
    const float* b4  = (const float*)d_in[17];
    const float* W5  = (const float*)d_in[18];
    const float* b5  = (const float*)d_in[19];
    float* out = (float*)d_out;

    // workspace layout (floats)
    float* w = (float*)d_ws;
    float* E1F  = w;                 // 65536
    float* E2F  = w + 65536;
    float* P1   = w + 131072;
    float* P2   = w + 196608;
    float* EMAT = w + 262144;        // 65536
    float* JACC = w + 327680;        // 256
    float* PART = w + 327936;        // 4096
    float* SCAL = w + 332032;        // 64
    unsigned* HIST = (unsigned*)(w + 332096); // 512
    float* INVN = w + 332608;        // 512
    bf16* bb   = (bf16*)(w + 333120);
    bf16* E1B  = bb;                 // 65536
    bf16* E2B  = bb + 65536;
    bf16* W3CT = bb + 131072;        // 65536
    bf16* W4T  = bb + 196608;        // 32768

    const int N_REC = 256 * 3 * 64 * 64;   // 3145728
    const int N_CF1 = 256 * 256 * 8 * 8;   // 4194304
    const int N_CF2 = 256 * 512 * 4 * 4;   // 2097152

    zero_hist_kernel<<<1, 512, 0, stream>>>(HIST);
    conv_weights_kernel<<<384, 256, 0, stream>>>(W3, W4, W3CT, W4T);

    encoder_kernel<<<256, 256, 0, stream>>>(f_orig,  W1, b1, g1, be1, W2, b2, E1F, E1B);
    encoder_kernel<<<256, 256, 0, stream>>>(f_recon, W1, b1, g1, be1, W2, b2, E2F, E2B);
    pmat_kernel<<<512, 256, 0, stream>>>(E1F, E2F, W3, b3, P1, P2);

    invnorm_kernel<<<512, 256, 0, stream>>>(f_orig, f_recon, INVN);
    topk_kernel<<<256, 256, 0, stream>>>(f_orig, f_recon, INVN, JACC);
    jacc_final_kernel<<<1, 256, 0, stream>>>(JACC, SCAL);

    recon_part_kernel<<<512, 256, 0, stream>>>(outputs, images, N_REC, PART);
    recon_final_kernel<<<1, 512, 0, stream>>>(PART, SCAL);

    stats_part_kernel<<<512, 256, 0, stream>>>(cf1, N_CF1, PART + 512, PART + 1024, PART + 1536);
    stats_part_kernel<<<512, 256, 0, stream>>>(cf2, N_CF2, PART + 2048, PART + 2560, PART + 3072);
    stats_final_kernel<<<1, 512, 0, stream>>>(PART + 512,  PART + 1024, PART + 1536, SCAL, 1, 3, 4);
    stats_final_kernel<<<1, 512, 0, stream>>>(PART + 2048, PART + 2560, PART + 3072, SCAL, 2, 5, 6);

    hist_kernel<<<512, 256, 0, stream>>>(cf1, N_CF1, SCAL, 3, 4, HIST);
    hist_kernel<<<512, 256, 0, stream>>>(cf2, N_CF2, SCAL, 5, 6, HIST + 256);
    entropy_kernel<<<1, 256, 0, stream>>>(HIST,       (float)N_CF1, SCAL, 7);
    entropy_kernel<<<1, 256, 0, stream>>>(HIST + 256, (float)N_CF2, SCAL, 8);

    energy_kernel<<<256, 128, 0, stream>>>(E1B, E2B, P1, P2, W3CT, W4T,
                                           g3, be3, b4, W5, b5, EMAT);

    margin_part_kernel<<<256, 256, 0, stream>>>(EMAT, PART + 3584);
    margin_final_kernel<<<1, 256, 0, stream>>>(PART + 3584, SCAL);

    combine_kernel<<<1, 1, 0, stream>>>(SCAL, out);
}
